// SelfAttention_44126493999188
// MI455X (gfx1250) — matrix operand order repaired and checked by reading, not compiled, not run on hardware
//
#include <hip/hip_runtime.h>
#include <hip/hip_bf16.h>
#include <math.h>

// ---------------------------------------------------------------------------
// Self-attention forward for MI455X (gfx1250), wave32 + WMMA f16->f32.
// B=4, T=1024, E=512, H=8 (per-head dim == E == 512).
// Pipeline: f32->f16 converts; 3 projection GEMMs (V stored transposed);
// per-(b,h) blocked attention (async-LDS Q staging, LDS score block,
// softmax); output GEMM with bias.
// ---------------------------------------------------------------------------

typedef _Float16 h16v __attribute__((ext_vector_type(16)));
typedef float    f32x8 __attribute__((ext_vector_type(8)));

union FragU { uint4 u[2]; h16v h; };

// A-matrix fragment, 16x32 f16, source row-major [M,K].
// ISA layout: lane l (m = l&15, g = l>>4) holds K = g*8+[0..7] and 16+g*8+[0..7].
__device__ __forceinline__ h16v load_a_frag(const _Float16* base, int ld, int lane) {
  const int r = lane & 15, g = lane >> 4;
  const _Float16* p = base + (size_t)r * ld + g * 8;
  FragU f;
  f.u[0] = *(const uint4*)(p);        // K = g*8 .. g*8+7
  f.u[1] = *(const uint4*)(p + 16);   // K = 16+g*8 .. 16+g*8+7
  return f.h;
}

// B-matrix fragment, 32x16 f16, source is B^T row-major [N,K] (K contiguous).
// Layout (per ISA 16-bit B pattern): lane l (n = l&15, g = l>>4) holds K = g*8+[0..7] and 16+g*8+[0..7].
__device__ __forceinline__ h16v load_bt_frag(const _Float16* base, int ld, int lane) {
  const int c = lane & 15, g = lane >> 4;
  const _Float16* p = base + (size_t)c * ld + g * 8;
  FragU f;
  f.u[0] = *(const uint4*)(p);        // K = g*8 .. g*8+7
  f.u[1] = *(const uint4*)(p + 16);   // K = 16+g*8 .. 16+g*8+7
  return f.h;
}

__device__ __forceinline__ f32x8 wmma_f16(h16v a, h16v b, f32x8 c) {
  return __builtin_amdgcn_wmma_f32_16x16x32_f16(false, a, false, b, (short)0, c,
                                                false, false);
}

// CDNA5 async memory->LDS copy (ASYNCcnt tracked), 16 bytes per lane.
// vdst = per-lane LDS byte offset, vaddr = per-lane 64-bit global address.
__device__ __forceinline__ void async_load_b128_to_lds(unsigned lds_off,
                                                       unsigned long long gaddr) {
  asm volatile("global_load_async_to_lds_b128 %0, %1, off"
               :: "v"(lds_off), "v"(gaddr) : "memory");
}
__device__ __forceinline__ void wait_asynccnt0() {
  asm volatile("s_wait_asynccnt 0" ::: "memory");
}

// ---------------------------------------------------------------------------
// f32 -> f16 convert
// ---------------------------------------------------------------------------
__global__ void cvt_f32_to_f16(const float* __restrict__ src,
                               _Float16* __restrict__ dst, int n) {
  for (int i = blockIdx.x * blockDim.x + threadIdx.x; i < n;
       i += gridDim.x * blockDim.x)
    dst[i] = (_Float16)src[i];
}

// ---------------------------------------------------------------------------
// Projection GEMM: out[m,n] = sum_k A[m,k] * Bt[n,k]
// A = x_f16 [4096,512], Bt = W_f16 [4096,512].
// Wave computes a 32x64 tile (2 A-frags x 4 B-frags, 8 accumulators).
// Output scattered to [b,h,t,e] (mode 0) or [b,h,e,t] (mode 1, for V).
// ---------------------------------------------------------------------------
__global__ __launch_bounds__(128) void proj_gemm_kernel(
    const _Float16* __restrict__ A, const _Float16* __restrict__ Bt,
    _Float16* __restrict__ out, int transpose_out) {
  constexpr int Tc = 1024, Hc = 8, Ec = 512, K = 512;
  const int lane = threadIdx.x & 31;
  const int wv = threadIdx.x >> 5;
  const int m0 = (blockIdx.x * 4 + wv) * 32;
  const int n0 = blockIdx.y * 64;

  f32x8 acc[2][4] = {};
  for (int kc = 0; kc < K; kc += 32) {
    h16v a0 = load_a_frag(A + (size_t)m0 * K + kc, K, lane);
    h16v a1 = load_a_frag(A + (size_t)(m0 + 16) * K + kc, K, lane);
#pragma unroll
    for (int t = 0; t < 4; ++t) {
      h16v b = load_bt_frag(Bt + (size_t)(n0 + t * 16) * K + kc, K, lane);
      acc[0][t] = wmma_f16(a0, b, acc[0][t]);
      acc[1][t] = wmma_f16(a1, b, acc[1][t]);
    }
  }

  const int c = lane & 15, g = lane >> 4;
#pragma unroll
  for (int aa = 0; aa < 2; ++aa) {
#pragma unroll
    for (int t = 0; t < 4; ++t) {
#pragma unroll
      for (int r = 0; r < 8; ++r) {
        const int m = m0 + aa * 16 + g * 8 + r;  // flat (b*T + t) row
        const int n = n0 + t * 16 + c;           // flat (h*E + e) col
        const int b = m >> 10, tq = m & (Tc - 1);
        const int h = n >> 9, e = n & (Ec - 1);
        const size_t idx =
            transpose_out ? ((size_t)((b * Hc + h) * Ec + e) * Tc + tq)
                          : ((size_t)((b * Hc + h) * Tc + tq) * Ec + e);
        out[idx] = (_Float16)acc[aa][t][r];
      }
    }
  }
}

// ---------------------------------------------------------------------------
// Attention: one workgroup (4 waves) per (16 query rows, h, b).
// Pass1: S = Q K^T * scale - mask  (WMMA, into LDS as f16)
// Pass2: row softmax stats (max, sum of exp), P = exp(S - max) in LDS
// Pass3: O = (P V) / rowsum  (WMMA, Vt is [b,h,e,t] so B^T rows contiguous)
// ---------------------------------------------------------------------------
__global__ __launch_bounds__(128) void attn_kernel(
    const _Float16* __restrict__ Qh, const _Float16* __restrict__ Kh,
    const _Float16* __restrict__ Vt, const float* __restrict__ mask,
    _Float16* __restrict__ Oc) {
  __shared__ _Float16 Qs[16][520];    // 16 x 512 (+8 pad) f16
  __shared__ _Float16 Ps[16][1032];   // 16 x 1024 (+8 pad) f16 scores -> exp
  __shared__ float red[16][8];
  __shared__ float rowmax[16];
  __shared__ float rowinv[16];

  const int tid = threadIdx.x, lane = tid & 31, wv = tid >> 5;
  const int h = blockIdx.y, bz = blockIdx.z;
  const int q0 = blockIdx.x * 16;
  const size_t bh = (size_t)(bz * 8 + h);
  const float scale = 0.04419417382415922f;  // 1/sqrt(512)

  // Stage Q block into LDS via CDNA5 async loads (no VGPR round-trip).
  // 16 rows x 512 halfs = 1024 x 16B transfers; 8 per thread.
  {
    const _Float16* Qg = Qh + (bh * 1024 + q0) * 512;
    const unsigned qs_base = (unsigned)(unsigned long long)&Qs[0][0];
#pragma unroll
    for (int j = 0; j < 8; ++j) {
      const int i = j * 128 + tid;           // 0..1023
      const int row = i >> 6, col = i & 63;  // 64 x 16B per row
      const unsigned lds_off = qs_base + (unsigned)(row * 520 * 2 + col * 16);
      const unsigned long long ga =
          (unsigned long long)(Qg + (size_t)row * 512 + col * 8);
      async_load_b128_to_lds(lds_off, ga);
    }
    wait_asynccnt0();
  }
  __syncthreads();

  // ---- Pass 1: scores ----
  for (int kt = wv; kt < 64; kt += 4) {      // 16 key-tiles per wave
    f32x8 acc = {};
    const _Float16* Kbase = Kh + (bh * 1024 + (size_t)kt * 16) * 512;
    for (int kc = 0; kc < 512; kc += 32) {
      h16v a = load_a_frag(&Qs[0][0] + kc, 520, lane);
      h16v b = load_bt_frag(Kbase + kc, 512, lane);
      acc = wmma_f16(a, b, acc);
    }
    const int c = lane & 15, g = lane >> 4;
    const int key = kt * 16 + c;
    const float* mrow = mask + (size_t)(bz * 1024 + q0) * 1024 + key;
#pragma unroll
    for (int r = 0; r < 8; ++r) {
      const int qrl = g * 8 + r;
      const float s = acc[r] * scale - mrow[(size_t)qrl * 1024];
      Ps[qrl][key] = (_Float16)s;
    }
  }
  __syncthreads();

  // ---- Pass 2: softmax stats over each of the 16 rows (8 threads/row) ----
  const int row = tid >> 3, sub = tid & 7;
  float mx = -3.0e38f;
  for (int j = sub * 128; j < sub * 128 + 128; ++j)
    mx = fmaxf(mx, (float)Ps[row][j]);
  red[row][sub] = mx;
  __syncthreads();
  if (sub == 0) {
    float m = red[row][0];
    for (int j2 = 1; j2 < 8; ++j2) m = fmaxf(m, red[row][j2]);
    rowmax[row] = m;
  }
  __syncthreads();
  const float rm = rowmax[row];
  float sum = 0.f;
  for (int j = sub * 128; j < sub * 128 + 128; ++j) {
    const float e = __expf((float)Ps[row][j] - rm);
    Ps[row][j] = (_Float16)e;
    sum += e;
  }
  __syncthreads();  // red[] free for reuse after all sub==0 reads done
  red[row][sub] = sum;
  __syncthreads();
  if (sub == 0) {
    float s = 0.f;
    for (int j2 = 0; j2 < 8; ++j2) s += red[row][j2];
    rowinv[row] = 1.0f / s;
  }
  __syncthreads();

  // ---- Pass 3: O = P V, each wave owns 128 of 512 e-columns ----
  for (int nt = 0; nt < 8; ++nt) {
    const int n0 = wv * 128 + nt * 16;
    f32x8 acc = {};
    const _Float16* Vb = Vt + (bh * 512 + n0) * 1024;   // [e,t] rows
    for (int kc = 0; kc < 1024; kc += 32) {
      h16v a = load_a_frag(&Ps[0][0] + kc, 1032, lane);
      h16v b = load_bt_frag(Vb + kc, 1024, lane);
      acc = wmma_f16(a, b, acc);
    }
    const int c = lane & 15, g = lane >> 4;
#pragma unroll
    for (int r = 0; r < 8; ++r) {
      const int qrl = g * 8 + r;
      const float o = acc[r] * rowinv[qrl];
      // Oc laid out [b,t, h*E+e] for the output GEMM
      Oc[(size_t)(bz * 1024 + q0 + qrl) * 4096 + h * 512 + n0 + c] =
          (_Float16)o;
    }
  }
}

// ---------------------------------------------------------------------------
// Output GEMM: out[m,n] = sum_k Oc[m,k] * Wu[n,k] + bu[n]
// Oc [4096,4096] f16, Wu_f16 [512,4096], out f32 [4096,512].
// Wave computes a 32x64 tile.
// ---------------------------------------------------------------------------
__global__ __launch_bounds__(128) void out_gemm_kernel(
    const _Float16* __restrict__ A, const _Float16* __restrict__ Bt,
    const float* __restrict__ bias, float* __restrict__ out) {
  constexpr int K = 4096, N = 512;
  const int lane = threadIdx.x & 31;
  const int wv = threadIdx.x >> 5;
  const int m0 = (blockIdx.x * 4 + wv) * 32;
  const int n0 = blockIdx.y * 64;

  f32x8 acc[2][4] = {};
  for (int kc = 0; kc < K; kc += 32) {
    h16v a0 = load_a_frag(A + (size_t)m0 * K + kc, K, lane);
    h16v a1 = load_a_frag(A + (size_t)(m0 + 16) * K + kc, K, lane);
#pragma unroll
    for (int t = 0; t < 4; ++t) {
      h16v b = load_bt_frag(Bt + (size_t)(n0 + t * 16) * K + kc, K, lane);
      acc[0][t] = wmma_f16(a0, b, acc[0][t]);
      acc[1][t] = wmma_f16(a1, b, acc[1][t]);
    }
  }

  const int c = lane & 15, g = lane >> 4;
#pragma unroll
  for (int aa = 0; aa < 2; ++aa) {
#pragma unroll
    for (int t = 0; t < 4; ++t) {
#pragma unroll
      for (int r = 0; r < 8; ++r) {
        const int m = m0 + aa * 16 + g * 8 + r;
        const int n = n0 + t * 16 + c;
        out[(size_t)m * N + n] = acc[aa][t][r] + bias[n];
      }
    }
  }
}

// ---------------------------------------------------------------------------
// Host launcher
// ---------------------------------------------------------------------------
extern "C" void kernel_launch(void* const* d_in, const int* in_sizes, int n_in,
                              void* d_out, int out_size, void* d_ws,
                              size_t ws_size, hipStream_t stream) {
  (void)in_sizes; (void)n_in; (void)out_size; (void)ws_size;
  // setup_inputs order: x, mask, Wk, Wq, Wv, Wu, bu, heads
  const float* x    = (const float*)d_in[0];
  const float* mask = (const float*)d_in[1];
  const float* Wk   = (const float*)d_in[2];
  const float* Wq   = (const float*)d_in[3];
  const float* Wv   = (const float*)d_in[4];
  const float* Wu   = (const float*)d_in[5];
  const float* bu   = (const float*)d_in[6];

  const size_t XN = (size_t)4096 * 512;          // 2M elems (x and each W)
  const size_t QN = (size_t)4 * 8 * 1024 * 512;  // 16M elems (per Q/K/V/Oc)

  _Float16* p = (_Float16*)d_ws;
  _Float16* xh  = p; p += XN;
  _Float16* Wqh = p; p += XN;
  _Float16* Wkh = p; p += XN;
  _Float16* Wvh = p; p += XN;
  _Float16* Wuh = p; p += XN;
  _Float16* Qh  = p; p += QN;
  _Float16* Kh  = p; p += QN;
  _Float16* Vt  = p; p += QN;   // [b,h,e,t]
  _Float16* Oc  = p; p += QN;   // [b*t, h*e]

  const int nconv = (int)XN;
  cvt_f32_to_f16<<<4096, 256, 0, stream>>>(x,  xh,  nconv);
  cvt_f32_to_f16<<<4096, 256, 0, stream>>>(Wq, Wqh, nconv);
  cvt_f32_to_f16<<<4096, 256, 0, stream>>>(Wk, Wkh, nconv);
  cvt_f32_to_f16<<<4096, 256, 0, stream>>>(Wv, Wvh, nconv);
  cvt_f32_to_f16<<<4096, 256, 0, stream>>>(Wu, Wuh, nconv);

  // Projections: M=4096 rows, N=4096 cols; block = 4 waves x 32 rows, 64 cols
  dim3 pgrid(32, 64);
  proj_gemm_kernel<<<pgrid, 128, 0, stream>>>(xh, Wqh, Qh, 0);
  proj_gemm_kernel<<<pgrid, 128, 0, stream>>>(xh, Wkh, Kh, 0);
  proj_gemm_kernel<<<pgrid, 128, 0, stream>>>(xh, Wvh, Vt, 1);

  // Attention: (T/16, H, B) workgroups of 128 threads
  attn_kernel<<<dim3(64, 8, 4), 128, 0, stream>>>(Qh, Kh, Vt, mask, Oc);

  // Output projection + bias: M=4096, N=512
  out_gemm_kernel<<<dim3(32, 8), 128, 0, stream>>>(Oc, Wuh, bu, (float*)d_out);
}
